// SAE_46102178955327
// MI455X (gfx1250) — compile-verified, matrix operand order checked by reading, not run on hardware
//
#include <hip/hip_runtime.h>
#include <hip/hip_bf16.h>

// ---------------------------------------------------------------------------
// SAE fused loss for MI455X (gfx1250, wave32, WMMA).
// Encode GEMM runs on v_wmma_f32_16x16x32_bf16 (fp32 accumulate) with a
// 32x32 per-wave register tile (2x2 WMMA tiles) so each A/B fragment feeds
// two WMMAs; everything else (top-k select, sparse decode, loss) stays fp32.
// ---------------------------------------------------------------------------

#define N_ROWS 8192   // B*D*L = 4*4*512
#define HDIM   512
#define MFEAT  4096
#define TOPK_K 64

typedef __attribute__((ext_vector_type(16))) __bf16 v16bf;
typedef __attribute__((ext_vector_type(8)))  __bf16 v8bf;
typedef __attribute__((ext_vector_type(8)))  float  v8f;

// ---------------- workspace layout (bytes) ----------------
// [0, 256)                    : 2 float accumulators (sse, sumz)
// [256, 256+4MB)              : enc as bf16 [MFEAT][HDIM]
// [.., +8MB)                  : decT fp32 [MFEAT][HDIM]  (dec transposed)
// [.., +128MB)                : dense z fp32 [N_ROWS][MFEAT]
#define WS_OFF_ENCBF  (256)
#define WS_OFF_DECT   (WS_OFF_ENCBF + (size_t)MFEAT * HDIM * 2)
#define WS_OFF_Z      (WS_OFF_DECT  + (size_t)MFEAT * HDIM * 4)

// ---------------------------------------------------------------------------
// Kernel 1: enc fp32 -> bf16 (RNE via __bf16 cast) + zero loss accumulators.
// ---------------------------------------------------------------------------
__global__ void sae_convert_enc(const float* __restrict__ enc,
                                __bf16* __restrict__ encbf,
                                float* __restrict__ acc) {
    if (blockIdx.x == 0 && threadIdx.x == 0) { acc[0] = 0.0f; acc[1] = 0.0f; }
    int i = blockIdx.x * blockDim.x + threadIdx.x;
    int stride = gridDim.x * blockDim.x;
    for (; i < MFEAT * HDIM; i += stride)
        encbf[i] = (__bf16)enc[i];
}

// ---------------------------------------------------------------------------
// Kernel 2: decT[j][h] = dec[h][j]  (fp32, LDS-tiled transpose)
// ---------------------------------------------------------------------------
__global__ void sae_transpose_dec(const float* __restrict__ dec,
                                  float* __restrict__ decT) {
    __shared__ float tile[32][33];
    int jb = blockIdx.x * 32;   // feature tile base (0..4095)
    int hb = blockIdx.y * 32;   // hidden tile base  (0..511)
    int tx = threadIdx.x;       // 0..31
    int ty = threadIdx.y;       // 0..7
    #pragma unroll
    for (int r = 0; r < 32; r += 8)
        tile[ty + r][tx] = dec[(size_t)(hb + ty + r) * MFEAT + (jb + tx)];
    __syncthreads();
    #pragma unroll
    for (int r = 0; r < 32; r += 8)
        decT[(size_t)(jb + ty + r) * HDIM + (hb + tx)] = tile[tx][ty + r];
}

// ---------------------------------------------------------------------------
// Kernel 3: encode GEMM + bias + ReLU -> dense z [N_ROWS][MFEAT] fp32.
// Grid: (MFEAT/256, N_ROWS/32). Block: 256 threads = 8 waves.
// WG covers 32 rows x 256 features. Each wave owns a 32x32 output tile:
// 2 row tiles x 2 feature tiles = 4 f32 accumulators; per 32-wide K step it
// loads 2 A frags (LDS) + 2 B frags (global bf16) and issues 4 WMMAs, so each
// fragment is reused twice (~1KB of loads per 16 KFLOP WMMA).
// ---------------------------------------------------------------------------
__global__ void __launch_bounds__(256)
sae_encode_gemm(const float* __restrict__ x,        // [N_ROWS][HDIM]
                const __bf16* __restrict__ encbf,   // [MFEAT][HDIM]
                const float* __restrict__ bias_pre, // [HDIM]
                const float* __restrict__ bias_enc, // [MFEAT]
                float* __restrict__ z) {            // [N_ROWS][MFEAT]
    __shared__ __bf16 ldsA[32 * HDIM];              // 32 KB

    const int tid     = threadIdx.x;
    const int rowBase = blockIdx.y * 32;
    const int wave    = tid >> 5;
    const int lane    = tid & 31;
    const int f0      = blockIdx.x * 256 + wave * 32;  // first 16-feat tile
    const int f1      = f0 + 16;                        // second 16-feat tile

    // Stage A: (x - bias_pre) -> bf16, row-major [32][512]. Coalesced in k.
    for (int i = tid; i < 32 * HDIM; i += 256) {
        int r = i >> 9;          // 0..31
        int k = i & (HDIM - 1);  // 0..511
        ldsA[i] = (__bf16)(x[(size_t)(rowBase + r) * HDIM + k] - bias_pre[k]);
    }
    __syncthreads();

    // WMMA 16-bit operand lane layout (symmetric for A row / B col):
    //   lanes 0-15 : row/col r, halves[0..7]=K+0..7,  halves[8..15]=K+16..23
    //   lanes16-31 : row/col r, halves[0..7]=K+8..15, halves[8..15]=K+24..31
    const int r    = lane & 15;
    const int koff = (lane >> 4) << 3;   // 0 or 8

    const __bf16* ldsA0 = &ldsA[r * HDIM + koff];           // rows 0..15
    const __bf16* ldsA1 = &ldsA[(16 + r) * HDIM + koff];    // rows 16..31
    const __bf16* brow0 = encbf + (size_t)(f0 + r) * HDIM + koff;
    const __bf16* brow1 = encbf + (size_t)(f1 + r) * HDIM + koff;

    v8f c00 = {}, c01 = {}, c10 = {}, c11 = {};
    #pragma unroll
    for (int k0 = 0; k0 < HDIM; k0 += 32) {
        v8bf t0, t1;

        t0 = *(const v8bf*)(ldsA0 + k0);
        t1 = *(const v8bf*)(ldsA0 + k0 + 16);
        v16bf a0 = __builtin_shufflevector(t0, t1,
                     0,1,2,3,4,5,6,7,8,9,10,11,12,13,14,15);
        t0 = *(const v8bf*)(ldsA1 + k0);
        t1 = *(const v8bf*)(ldsA1 + k0 + 16);
        v16bf a1 = __builtin_shufflevector(t0, t1,
                     0,1,2,3,4,5,6,7,8,9,10,11,12,13,14,15);

        t0 = *(const v8bf*)(brow0 + k0);
        t1 = *(const v8bf*)(brow0 + k0 + 16);
        v16bf b0 = __builtin_shufflevector(t0, t1,
                     0,1,2,3,4,5,6,7,8,9,10,11,12,13,14,15);
        t0 = *(const v8bf*)(brow1 + k0);
        t1 = *(const v8bf*)(brow1 + k0 + 16);
        v16bf b1 = __builtin_shufflevector(t0, t1,
                     0,1,2,3,4,5,6,7,8,9,10,11,12,13,14,15);

        c00 = __builtin_amdgcn_wmma_f32_16x16x32_bf16(
                  false, a0, false, b0, (short)0, c00, false, false);
        c01 = __builtin_amdgcn_wmma_f32_16x16x32_bf16(
                  false, a0, false, b1, (short)0, c01, false, false);
        c10 = __builtin_amdgcn_wmma_f32_16x16x32_bf16(
                  false, a1, false, b0, (short)0, c10, false, false);
        c11 = __builtin_amdgcn_wmma_f32_16x16x32_bf16(
                  false, a1, false, b1, (short)0, c11, false, false);
    }

    // C layout: VGPR i -> row M=i (lanes 0-15) or M=8+i (lanes 16-31), col=lane&15
    const int col0  = f0 + r;
    const int col1  = f1 + r;
    const float bb0 = bias_enc[col0];
    const float bb1 = bias_enc[col1];
    const int rbA = rowBase + ((lane >> 4) << 3);        // row tile 0
    const int rbB = rbA + 16;                            // row tile 1
    #pragma unroll
    for (int i = 0; i < 8; ++i) {
        float v;
        v = c00[i] + bb0; v = v > 0.0f ? v : 0.0f;
        z[(size_t)(rbA + i) * MFEAT + col0] = v;
        v = c01[i] + bb1; v = v > 0.0f ? v : 0.0f;
        z[(size_t)(rbA + i) * MFEAT + col1] = v;
        v = c10[i] + bb0; v = v > 0.0f ? v : 0.0f;
        z[(size_t)(rbB + i) * MFEAT + col0] = v;
        v = c11[i] + bb1; v = v > 0.0f ? v : 0.0f;
        z[(size_t)(rbB + i) * MFEAT + col1] = v;
    }
}

// ---------------------------------------------------------------------------
// Kernel 4: per-row exact top-64 (4-pass radix select on IEEE bits; all
// values >= 0 after ReLU so uint order == float order), sparse fp32 decode
// against decT, squared-error + L1 partial sums -> global atomics.
// Grid: N_ROWS blocks of 256 threads.
// ---------------------------------------------------------------------------
__global__ void __launch_bounds__(256)
sae_topk_decode_loss(const float* __restrict__ z,     // [N_ROWS][MFEAT]
                     const float* __restrict__ decT,  // [MFEAT][HDIM]
                     const float* __restrict__ xsrc,  // [N_ROWS][HDIM]
                     float* __restrict__ acc) {       // [sse, sumz]
    __shared__ float    zrow[MFEAT];      // 16 KB
    __shared__ unsigned hist[256];
    __shared__ unsigned bc_prefix, bc_kneed;
    __shared__ unsigned cnt_gt, sel_n, eq_n;
    __shared__ int      sel_idx[TOPK_K];
    __shared__ float    sel_val[TOPK_K];
    __shared__ float    lds_sse, lds_sumz;

    const int row = blockIdx.x;
    const int tid = threadIdx.x;

    for (int i = tid; i < MFEAT; i += 256)
        zrow[i] = z[(size_t)row * MFEAT + i];
    if (tid == 0) { lds_sse = 0.0f; lds_sumz = 0.0f; cnt_gt = 0; sel_n = 0; eq_n = 0; }
    __syncthreads();

    // ---- radix select: find bit pattern of the 64th-largest value ----
    unsigned prefix = 0;
    int kneed = TOPK_K;
    for (int shift = 24; shift >= 0; shift -= 8) {
        hist[tid] = 0;
        __syncthreads();
        unsigned hmask = (shift == 24) ? 0u : (0xFFFFFFFFu << (shift + 8));
        for (int i = tid; i < MFEAT; i += 256) {
            unsigned u = __float_as_uint(zrow[i]);
            if ((u & hmask) == prefix)
                atomicAdd(&hist[(u >> shift) & 255], 1u);
        }
        __syncthreads();
        if (tid == 0) {
            unsigned run = 0;
            int b = 255;
            for (; b > 0; --b) {
                unsigned h = hist[b];
                if (run + h >= (unsigned)kneed) break;
                run += h;
            }
            bc_prefix = prefix | ((unsigned)b << shift);
            bc_kneed  = (unsigned)kneed - run;
        }
        __syncthreads();
        prefix = bc_prefix;
        kneed  = (int)bc_kneed;
        __syncthreads();
    }
    const unsigned T = prefix;   // bit pattern of k-th largest value

    // ---- count strictly-greater, then gather top-64 (ties filled in) ----
    {
        unsigned local = 0;
        for (int i = tid; i < MFEAT; i += 256)
            if (__float_as_uint(zrow[i]) > T) local++;
        if (local) atomicAdd(&cnt_gt, local);
    }
    __syncthreads();
    const unsigned need_eq = TOPK_K - cnt_gt;
    for (int i = tid; i < MFEAT; i += 256) {
        unsigned u = __float_as_uint(zrow[i]);
        if (u > T) {
            unsigned p = atomicAdd(&sel_n, 1u);
            sel_idx[p] = i; sel_val[p] = zrow[i];
        } else if (u == T) {
            unsigned e = atomicAdd(&eq_n, 1u);
            if (e < need_eq) {
                unsigned p = atomicAdd(&sel_n, 1u);
                sel_idx[p] = i; sel_val[p] = zrow[i];
            }
        }
    }
    __syncthreads();

    if (tid < TOPK_K) atomicAdd(&lds_sumz, sel_val[tid]);

    // ---- sparse decode (fp32) + squared error ----
    const int h0 = tid, h1 = tid + 256;
    float a0 = 0.0f, a1 = 0.0f;
    #pragma unroll 8
    for (int s = 0; s < TOPK_K; ++s) {
        float v = sel_val[s];
        const float* dr = decT + (size_t)sel_idx[s] * HDIM;
        a0 += v * dr[h0];
        a1 += v * dr[h1];
    }
    float d0 = a0 - xsrc[(size_t)row * HDIM + h0];
    float d1 = a1 - xsrc[(size_t)row * HDIM + h1];
    atomicAdd(&lds_sse, d0 * d0 + d1 * d1);
    __syncthreads();
    if (tid == 0) {
        atomicAdd(&acc[0], lds_sse);
        atomicAdd(&acc[1], lds_sumz);
    }
}

// ---------------------------------------------------------------------------
// Kernel 5: finalize scalar loss (d_out is NOT assumed pre-zeroed).
// ---------------------------------------------------------------------------
__global__ void sae_finalize(const float* __restrict__ acc, float* __restrict__ out) {
    if (threadIdx.x == 0 && blockIdx.x == 0) {
        float recon  = acc[0] / ((float)N_ROWS * (float)HDIM);
        float sparse = acc[1] / ((float)N_ROWS * (float)MFEAT);
        out[0] = recon + 1e-3f * sparse;
    }
}

// ---------------------------------------------------------------------------
extern "C" void kernel_launch(void* const* d_in, const int* in_sizes, int n_in,
                              void* d_out, int out_size, void* d_ws, size_t ws_size,
                              hipStream_t stream) {
    const float* zL       = (const float*)d_in[0];  // [4,4,512,512] -> [8192,512]
    const float* enc      = (const float*)d_in[1];  // [4096,512]
    const float* dec      = (const float*)d_in[2];  // [512,4096]
    const float* bias_pre = (const float*)d_in[3];  // [512]
    const float* bias_enc = (const float*)d_in[4];  // [4096]
    float* out = (float*)d_out;

    char* ws = (char*)d_ws;
    float*  acc   = (float*)ws;
    __bf16* encbf = (__bf16*)(ws + WS_OFF_ENCBF);
    float*  decT  = (float*)(ws + WS_OFF_DECT);
    float*  zbuf  = (float*)(ws + WS_OFF_Z);

    // 1) enc -> bf16, zero accumulators
    sae_convert_enc<<<2048, 256, 0, stream>>>(enc, encbf, acc);

    // 2) dec -> decT (fp32, coalesced decode reads later)
    sae_transpose_dec<<<dim3(MFEAT / 32, HDIM / 32), dim3(32, 8), 0, stream>>>(dec, decT);

    // 3) encode GEMM (bf16 WMMA, fp32 accum, 32x32 per-wave tile) -> dense z
    sae_encode_gemm<<<dim3(MFEAT / 256, N_ROWS / 32), 256, 0, stream>>>(
        zL, encbf, bias_pre, bias_enc, zbuf);

    // 4) per-row top-64 + sparse decode + loss partials
    sae_topk_decode_loss<<<N_ROWS, 256, 0, stream>>>(zbuf, decT, zL, acc);

    // 5) scalar loss
    sae_finalize<<<1, 1, 0, stream>>>(acc, out);
}
